// LSTM_Cell_23553600651695
// MI455X (gfx1250) — compile-verified, matrix operand order checked by reading, not run on hardware
//
#include <hip/hip_runtime.h>
#include <hip/hip_bf16.h>

typedef __attribute__((ext_vector_type(16))) __bf16 v16bf;
typedef __attribute__((ext_vector_type(8)))  float  v8f;

using u16 = unsigned short;
using u32 = unsigned int;

#define T_STEPS 2048
#define BATCH   64
#define DIN     512
#define HID     512
#define NWG     32          // 32 persistent WGs (2 share a column slice), 1 wave/SIMD
#define THREADS 128         // 4 waves of 32

// ---- workspace layout (bytes) ----
#define WS_COUNTER 0            // 256 B   (grid barrier counter)
#define WS_BIAS    256          // 2048 f32 = 8192 B (bx+bh combined)
#define WS_HBUF0   8448         // 64x512 bf16 = 65536 B (131072 B slot)
#define WS_HBUF1   139520       // 64x512 bf16 = 65536 B (131072 B slot)
#define WS_WPACK   270592       // packed bf16 weights, 1M dwords = 4 MB
#define WS_XBF16   4464896      // X converted to bf16: 64M elems = 128 MB
#define X_ELEMS    ((size_t)T_STEPS * BATCH * DIN)      // 67,108,864
#define WS_NEEDED  (WS_XBF16 + X_ELEMS * 2)

__device__ __forceinline__ u16 f2bf(float f) {          // RNE
    u32 u = __float_as_uint(f);
    return (u16)((u + 0x7FFFu + ((u >> 16) & 1u)) >> 16);
}
__device__ __forceinline__ u32 pk2(float a, float b) {  // RNE pair pack
    return (u32)f2bf(a) | ((u32)f2bf(b) << 16);
}
__device__ __forceinline__ u32 pk2t(float a, float b) { // truncation pair pack (cheap)
    return (__float_as_uint(a) >> 16) | (__float_as_uint(b) & 0xffff0000u);
}
__device__ __forceinline__ float fast_sigmoid(float x) {
    return __builtin_amdgcn_rcpf(1.f + __expf(-x));     // v_exp + v_rcp
}
__device__ __forceinline__ float fast_tanh(float x) {
    return 1.f - 2.f * __builtin_amdgcn_rcpf(1.f + __expf(2.f * x));
}

union ABTile { v16bf v; u32 u[8]; uint4 q[2]; };

// 4 gate-tile WMMAs sharing one A tile. B tiles for one k-tile are packed in
// one contiguous 8KB LDS block: [gate*2+jt][lane][8 dwords] -> all 8
// ds_load_b128 use 16-bit offset immediates off a single per-k-tile base.
__device__ __forceinline__ void mma4(const ABTile& a, const u32* lp,
                                     v8f& a0, v8f& a1, v8f& a2, v8f& a3) {
    ABTile b0, b1, b2, b3;
    b0.q[0] = *(const uint4*)(lp);        b0.q[1] = *(const uint4*)(lp + 4);
    b1.q[0] = *(const uint4*)(lp + 512);  b1.q[1] = *(const uint4*)(lp + 516);
    b2.q[0] = *(const uint4*)(lp + 1024); b2.q[1] = *(const uint4*)(lp + 1028);
    b3.q[0] = *(const uint4*)(lp + 1536); b3.q[1] = *(const uint4*)(lp + 1540);
    a0 = __builtin_amdgcn_wmma_f32_16x16x32_bf16(false, a.v, false, b0.v, (short)0, a0, false, false);
    a1 = __builtin_amdgcn_wmma_f32_16x16x32_bf16(false, a.v, false, b1.v, (short)0, a1, false, false);
    a2 = __builtin_amdgcn_wmma_f32_16x16x32_bf16(false, a.v, false, b2.v, (short)0, a2, false, false);
    a3 = __builtin_amdgcn_wmma_f32_16x16x32_bf16(false, a.v, false, b3.v, (short)0, a3, false, false);
}

// X-part: 16 k-tiles from pre-converted bf16 X
__device__ __forceinline__ void x_part_pre(const u16* __restrict__ Xt, const u32* ldsLane,
                                           v8f& a0, v8f& a1, v8f& a2, v8f& a3) {
    #pragma unroll 2
    for (u32 kt = 0; kt < 16u; ++kt) {
        const u16* p = Xt + kt * 32u;
        ABTile a;
        a.q[0] = *(const uint4*)(p);
        a.q[1] = *(const uint4*)(p + 16);
        mma4(a, ldsLane + kt * 2048u, a0, a1, a2, a3);
    }
}

// X-part fallback: fp32 X, cheap truncation pack
__device__ __forceinline__ void x_part_raw(const float* __restrict__ Xt, const u32* ldsLane,
                                           v8f& a0, v8f& a1, v8f& a2, v8f& a3) {
    #pragma unroll 2
    for (u32 kt = 0; kt < 16u; ++kt) {
        const float* p = Xt + kt * 32u;
        float4 x0 = *(const float4*)(p);
        float4 x1 = *(const float4*)(p + 4);
        float4 x2 = *(const float4*)(p + 16);
        float4 x3 = *(const float4*)(p + 20);
        ABTile a;
        a.u[0] = pk2t(x0.x, x0.y); a.u[1] = pk2t(x0.z, x0.w);
        a.u[2] = pk2t(x1.x, x1.y); a.u[3] = pk2t(x1.z, x1.w);
        a.u[4] = pk2t(x2.x, x2.y); a.u[5] = pk2t(x2.z, x2.w);
        a.u[6] = pk2t(x3.x, x3.y); a.u[7] = pk2t(x3.z, x3.w);
        mma4(a, ldsLane + kt * 2048u, a0, a1, a2, a3);
    }
}

// h-part: 16 k-tiles (LDS k-blocks 16..31) -- the serial-critical half.
// Deeper unroll: more DS loads in flight ahead of the WMMA chain.
__device__ __forceinline__ void h_part(const u16* __restrict__ Ht, const u32* ldsLane,
                                       v8f& a0, v8f& a1, v8f& a2, v8f& a3) {
    #pragma unroll 4
    for (u32 kt = 0; kt < 16u; ++kt) {
        const u16* p = Ht + kt * 32u;
        ABTile a;
        a.q[0] = *(const uint4*)(p);
        a.q[1] = *(const uint4*)(p + 16);
        mma4(a, ldsLane + (kt + 16u) * 2048u, a0, a1, a2, a3);
    }
}

// ------------------------------------------------------------------
// Prep 1: combined bias  bias[n] = bx[n] + bh[n]
// ------------------------------------------------------------------
__global__ void make_bias_kernel(const float* __restrict__ bx,
                                 const float* __restrict__ bh,
                                 float* __restrict__ bias) {
    int n = blockIdx.x * blockDim.x + threadIdx.x;
    if (n < 4 * HID) bias[n] = bx[n] + bh[n];
}

// ------------------------------------------------------------------
// Prep 2: X fp32 -> bf16 (RNE), bandwidth-bound one-time pass.
// ------------------------------------------------------------------
__global__ void convert_x_kernel(const float* __restrict__ X, u16* __restrict__ Xb) {
    size_t i = ((size_t)blockIdx.x * blockDim.x + threadIdx.x) * 8;
    float4 x0 = *(const float4*)(X + i);
    float4 x1 = *(const float4*)(X + i + 4);
    uint4 o;
    o.x = pk2(x0.x, x0.y); o.y = pk2(x0.z, x0.w);
    o.z = pk2(x1.x, x1.y); o.w = pk2(x1.z, x1.w);
    *(uint4*)(Xb + i) = o;
}

// ------------------------------------------------------------------
// Prep 3: pack [Wx; Wh] -> bf16 WMMA B-tile layout.
// Blob order: [w 0..15][kt 0..31][lt = gate*2+jt 0..7][lane][dword]
// (k-tile-major so each k-tile's 8 tiles are one contiguous 8KB block)
// ------------------------------------------------------------------
__global__ void pack_weights_kernel(const float* __restrict__ Wx,
                                    const float* __restrict__ Wh,
                                    u32* __restrict__ Wp) {
    u32 i = blockIdx.x * blockDim.x + threadIdx.x;    // < 1<<20
    u32 d    = i & 7u;
    u32 lane = (i >> 3) & 31u;
    u32 lt   = (i >> 8) & 7u;
    u32 kt   = (i >> 11) & 31u;
    u32 w    = (i >> 16) & 15u;
    u32 g  = lt >> 1, jt = lt & 1u;
    u32 n  = g * HID + w * 32u + jt * 16u + (lane & 15u);
    u32 kb = (lane >> 4) * 8u;
    u32 kit = (d < 4u) ? (kb + 2u * d) : (kb + 16u + 2u * (d - 4u));
    u32 k = kt * 32u + kit;                           // 0..1023, even
    float f0, f1;
    if (k < DIN) { f0 = Wx[k * 2048u + n];          f1 = Wx[(k + 1u) * 2048u + n]; }
    else { u32 kk = k - DIN; f0 = Wh[kk * 2048u + n]; f1 = Wh[(kk + 1u) * 2048u + n]; }
    Wp[i] = pk2(f0, f1);
}

// ------------------------------------------------------------------
// Persistent LSTM: 32 WGs x 4 waves (1 wave/SIMD).
// Block (mh, w): batch rows [mh*32, +32), column slice w (32 cols x 4 gates).
// Wave (mt_l, jt): 4 gate tiles for rows [(mh*2+mt_l)*16,+16) x cols
// [w*32+jt*16,+16). X-part of step t+1 is computed BEFORE the step-t
// barrier wait, so only the h-part + gates sit on the serial chain.
// ------------------------------------------------------------------
__global__ void __launch_bounds__(THREADS)
lstm_persistent_kernel(const float* __restrict__ X,
                       const u16*   __restrict__ Xb,     // may be unused
                       const u32*   __restrict__ Wp,
                       const float* __restrict__ bias,
                       u16* __restrict__ h0b, u16* __restrict__ h1b,
                       u32* __restrict__ counter,
                       float* __restrict__ out,
                       int usePre) {
    extern __shared__ u32 sW[];                       // 65536 dwords = 256 KB

    const u32 w    = blockIdx.x & 15u;                // column slice
    const u32 mh   = blockIdx.x >> 4;                 // batch half 0..1
    const u32 tid  = threadIdx.x;
    const u32 lane = tid & 31u;
    const u32 wave = tid >> 5;        // 0..3
    const u32 mt   = mh * 2u + (wave >> 1);           // batch tile 0..3
    const u32 jt   = wave & 1u;                       // column sub-tile 0..1

    // ---- preload this column slice's 256KB weights into LDS ----
    {
        const uint4* src = (const uint4*)(Wp + (size_t)w * 65536u);
        uint4* dst = (uint4*)sW;
        #pragma unroll 4
        for (u32 idx = tid; idx < 16384u; idx += THREADS) dst[idx] = src[idx];
    }
    __syncthreads();

    const u32 m_lo  = lane & 15u;
    const u32 khalf = (lane >> 4) * 8u;                 // A/B K-half select
    const u32 b0    = mt * 16u + m_lo;                  // A-load row (batch)
    const u32 jcol  = w * 32u + jt * 16u + m_lo;        // output column j
    const u32 brow  = mt * 16u + ((lane >> 4) << 3);    // C-layout row base

    const float bi = bias[jcol];
    const float bf = bias[HID + jcol];
    const float bo = bias[2 * HID + jcol];
    const float bz = bias[3 * HID + jcol];

    float cst[8], hs[8], cs[8];
    #pragma unroll
    for (int e = 0; e < 8; ++e) { cst[e] = 0.f; hs[e] = 0.f; cs[e] = 0.f; }

    const u32* ldsLane = sW + jt * 256u + lane * 8u;    // + kt*2048 + g*512
    const size_t aoff = (size_t)b0 * DIN + khalf;       // per-lane A offset

    // ---- prologue: X-part of step 0 ----
    v8f xa0 = {0,0,0,0,0,0,0,0}, xa1 = {0,0,0,0,0,0,0,0};
    v8f xa2 = {0,0,0,0,0,0,0,0}, xa3 = {0,0,0,0,0,0,0,0};
    if (usePre) x_part_pre(Xb + aoff, ldsLane, xa0, xa1, xa2, xa3);
    else        x_part_raw(X  + aoff, ldsLane, xa0, xa1, xa2, xa3);

    for (int t = 0; t < T_STEPS; ++t) {
        const u16* hc = (t & 1) ? h1b : h0b;
        u16*       hn = (t & 1) ? h0b : h1b;

        // ---- serial-critical: h-part on top of the X-partials ----
        v8f a0 = xa0, a1 = xa1, a2 = xa2, a3 = xa3;
        h_part(hc + aoff, ldsLane, a0, a1, a2, a3);

        // ---- gates (i,f,o,z element-aligned across the 4 accumulators) ----
        #pragma unroll
        for (int e = 0; e < 8; ++e) {
            float iv = fast_sigmoid(a0[e] + bi);
            float fv = fast_sigmoid(a1[e] + bf);
            float ov = fast_sigmoid(a2[e] + bo);
            float zv = fast_tanh(a3[e] + bz);
            float c  = iv * zv + fv * cst[e];
            float h  = ov * fast_tanh(c);
            cst[e] = c;
            cs[e] += c;
            hs[e] += h;
            hn[(size_t)(brow + e) * HID + jcol] = f2bf(h);
        }

        // ---- off-chain: X-part of step t+1 (overlaps barrier latency) ----
        if (t + 1 < T_STEPS) {
            xa0 = (v8f){0,0,0,0,0,0,0,0}; xa1 = (v8f){0,0,0,0,0,0,0,0};
            xa2 = (v8f){0,0,0,0,0,0,0,0}; xa3 = (v8f){0,0,0,0,0,0,0,0};
            const size_t xo = (size_t)(t + 1) * (BATCH * DIN) + aoff;
            if (usePre) x_part_pre(Xb + xo, ldsLane, xa0, xa1, xa2, xa3);
            else        x_part_raw(X  + xo, ldsLane, xa0, xa1, xa2, xa3);
        }

        // ---- grid barrier (release h stores / acquire next h reads) ----
        __syncthreads();
        if (tid == 0) {
            __hip_atomic_fetch_add(counter, 1u, __ATOMIC_RELEASE, __HIP_MEMORY_SCOPE_AGENT);
            const u32 target = (u32)(t + 1) * NWG;
            while (__hip_atomic_load(counter, __ATOMIC_ACQUIRE, __HIP_MEMORY_SCOPE_AGENT) < target)
                __builtin_amdgcn_s_sleep(1);
        }
        __syncthreads();
    }

    // ---- means: out[0:32768) = h_sum/T, out[32768:65536) = c_sum/T ----
    const float invT = 1.0f / (float)T_STEPS;
    #pragma unroll
    for (int e = 0; e < 8; ++e) {
        out[(size_t)(brow + e) * HID + jcol]          = hs[e] * invT;
        out[32768u + (size_t)(brow + e) * HID + jcol] = cs[e] * invT;
    }
}

extern "C" void kernel_launch(void* const* d_in, const int* in_sizes, int n_in,
                              void* d_out, int out_size, void* d_ws, size_t ws_size,
                              hipStream_t stream) {
    const float* X  = (const float*)d_in[0];
    const float* Wx = (const float*)d_in[1];
    const float* bx = (const float*)d_in[2];
    const float* Wh = (const float*)d_in[3];
    const float* bh = (const float*)d_in[4];
    float* out = (float*)d_out;

    char* ws = (char*)d_ws;
    u32*   counter = (u32*)(ws + WS_COUNTER);
    float* bias    = (float*)(ws + WS_BIAS);
    u16*   h0b     = (u16*)(ws + WS_HBUF0);
    u16*   h1b     = (u16*)(ws + WS_HBUF1);
    u32*   Wp      = (u32*)(ws + WS_WPACK);
    u16*   Xb      = (u16*)(ws + WS_XBF16);

    const int usePre = (ws_size >= WS_NEEDED) ? 1 : 0;   // fixed per run -> deterministic

    hipMemsetAsync(counter, 0, 256, stream);
    hipMemsetAsync(h0b, 0, (size_t)BATCH * HID * sizeof(u16), stream);

    make_bias_kernel<<<8, 256, 0, stream>>>(bx, bh, bias);
    pack_weights_kernel<<<4096, 256, 0, stream>>>(Wx, Wh, Wp);
    if (usePre) {
        convert_x_kernel<<<32768, 256, 0, stream>>>(X, Xb);   // 8 elems/thread, exact
    }

    (void)hipFuncSetAttribute((const void*)lstm_persistent_kernel,
                              hipFuncAttributeMaxDynamicSharedMemorySize, 262144);
    lstm_persistent_kernel<<<NWG, THREADS, 262144, stream>>>(X, Xb, Wp, bias,
                                                             h0b, h1b, counter, out,
                                                             usePre);
}
